// MultiHeadAttention_57990648430611
// MI455X (gfx1250) — matrix operand order repaired and checked by reading, not compiled, not run on hardware
//
#include <hip/hip_runtime.h>
#include <hip/hip_bf16.h>

// ---------------- CDNA5 WMMA plumbing (gfx1250, wave32) ----------------
typedef __attribute__((ext_vector_type(16))) __bf16 v16bf;
typedef __attribute__((ext_vector_type(8)))  float  v8f;
typedef __attribute__((ext_vector_type(4)))  unsigned int u32x4;
typedef __attribute__((ext_vector_type(8)))  unsigned int u32x8;

union Frag16 { v16bf h; unsigned int u[8]; };

__device__ __forceinline__ v8f wmma_bf16(const Frag16& a, const Frag16& b, v8f c) {
    // D = A(16x32 bf16) x B(32x16 bf16) + C(16x16 f32)
    return __builtin_amdgcn_wmma_f32_16x16x32_bf16(
        false, a.h, false, b.h, (short)0, c, false, false);
}

// A-fragment (16x32 bf16). ISA layout: lane L (m = L&15, h = L>>4) holds
// K = 8h..8h+7 in u[0..3] and K = 16+8h..16+8h+7 in u[4..7] -> two 16B runs.
__device__ __forceinline__ void frag_a(Frag16& f, const unsigned short* p,
                                       int lda, int r0, int k0) {
    const int lane = threadIdx.x & 31;
    const unsigned short* q = p + (r0 + (lane & 15)) * lda + k0 + ((lane >> 4) << 3);
    uint4 x0 = *(const uint4*)(q);
    uint4 x1 = *(const uint4*)(q + 16);
    f.u[0] = x0.x; f.u[1] = x0.y; f.u[2] = x0.z; f.u[3] = x0.w;
    f.u[4] = x1.x; f.u[5] = x1.y; f.u[6] = x1.z; f.u[7] = x1.w;
}

// B-fragment (32x16 bf16) loaded from a tile stored TRANSPOSED in LDS, i.e.
// tile[n][k] = B[k][n]. Lane L (n = L&15, h = L>>4) needs K = 8h..8h+7 and
// K = 16+8h..16+8h+7 for column n -> two 16B runs.
__device__ __forceinline__ void frag_b(Frag16& f, const unsigned short* p,
                                       int lda, int c0, int k0) {
    const int lane = threadIdx.x & 31;
    const unsigned short* q = p + (c0 + (lane & 15)) * lda + k0 + ((lane >> 4) << 3);
    uint4 x0 = *(const uint4*)(q);
    uint4 x1 = *(const uint4*)(q + 16);
    f.u[0] = x0.x; f.u[1] = x0.y; f.u[2] = x0.z; f.u[3] = x0.w;
    f.u[4] = x1.x; f.u[5] = x1.y; f.u[6] = x1.z; f.u[7] = x1.w;
}

__device__ __forceinline__ unsigned short cvt1bf(float x) {  // f32 -> bf16 RNE
    unsigned int u = __float_as_uint(x);
    u = (u + 0x7FFFu + ((u >> 16) & 1u)) >> 16;
    return (unsigned short)u;
}
__device__ __forceinline__ unsigned int cvt2bf(float a, float b) {
    return (unsigned int)cvt1bf(a) | ((unsigned int)cvt1bf(b) << 16);
}

// ---------------- Tensor Data Mover (ISA ch.8 D# descriptor) -------------
// Group0: [1:0]=count=1, [63:32]=lds_addr, [120:64]=global_addr, [127:126]=2
__device__ __forceinline__ u32x4 tdm_g0(const void* gaddr, unsigned int lds_off) {
    unsigned long long ga = (unsigned long long)(size_t)gaddr;
    u32x4 g;
    g[0] = 1u;
    g[1] = lds_off;
    g[2] = (unsigned int)ga;
    g[3] = (unsigned int)(ga >> 32) | 0x80000000u;   // type=2 ("image")
    return g;
}
// Group1: data_size=2B; LDS padding 4 DWORDs per 32 DWORDs (64 bf16 row -> stride 72)
__device__ __forceinline__ u32x8 tdm_g1(unsigned int td0, unsigned int td1,
                                        unsigned int tl0, unsigned int tl1,
                                        unsigned long long s0) {
    u32x8 g;
    g[0] = 0x10000u | (1u << 20) | (4u << 22) | (3u << 25);
    g[1] = (td0 & 0xFFFFu) << 16;
    g[2] = (td0 >> 16) | ((td1 & 0xFFFFu) << 16);
    g[3] = (td1 >> 16) | (tl0 << 16);
    g[4] = tl1 & 0xFFFFu;
    g[5] = (unsigned int)s0;
    g[6] = (unsigned int)(s0 >> 32) & 0xFFFFu;
    g[7] = 0u;
    return g;
}
__device__ __forceinline__ void tdm_load(u32x4 g0, u32x8 g1) {
    asm volatile("tensor_load_to_lds %0, %1" :: "s"(g0), "s"(g1) : "memory");
}

#define DMODEL 1024
#define NHEAD  16

// ---------------- GEMM: out[m,n] = sum_d X[m,d] * W[n,d] ----------------
// mode 0: bf16 out, head-split  [b,h,s,dh]
// mode 1: bf16 out, head-split transposed [b,h,dh,s]   (for V)
// mode 2: f32 out = acc + resid[m,n] -> out_f[m*D+n]   (output projection)
__global__ __launch_bounds__(256)
void gemm_xwT(const float* __restrict__ X, const float* __restrict__ W,
              unsigned short* __restrict__ out_bf, float* __restrict__ out_f,
              const float* __restrict__ resid, int S, int mode) {
    __shared__ unsigned short Xs[64 * 40];
    __shared__ unsigned short Ws[64 * 40];
    const int tid = threadIdx.x;
    const int wave = tid >> 5, lane = tid & 31;
    const int wm = wave >> 1, wn = wave & 1;
    const int m0 = blockIdx.y * 64;
    const int n0 = blockIdx.x * 64;
    v8f acc0 = {}, acc1 = {};

    for (int k0 = 0; k0 < DMODEL; k0 += 32) {
        for (int i = tid; i < 512; i += 256) {           // 64x32 f32 -> bf16
            int r = i >> 3, c4 = (i & 7) * 4;
            float4 fx = *(const float4*)(X + (size_t)(m0 + r) * DMODEL + k0 + c4);
            *(unsigned int*)(Xs + r * 40 + c4)     = cvt2bf(fx.x, fx.y);
            *(unsigned int*)(Xs + r * 40 + c4 + 2) = cvt2bf(fx.z, fx.w);
            float4 fw = *(const float4*)(W + (size_t)(n0 + r) * DMODEL + k0 + c4);
            *(unsigned int*)(Ws + r * 40 + c4)     = cvt2bf(fw.x, fw.y);
            *(unsigned int*)(Ws + r * 40 + c4 + 2) = cvt2bf(fw.z, fw.w);
        }
        __syncthreads();
        Frag16 a, b0, b1;
        frag_a(a,  Xs, 40, wm * 16, 0);
        frag_b(b0, Ws, 40, wn * 32, 0);
        frag_b(b1, Ws, 40, wn * 32 + 16, 0);
        acc0 = wmma_bf16(a, b0, acc0);
        acc1 = wmma_bf16(a, b1, acc1);
        __syncthreads();
    }

    const int hh = lane >> 4, ln = lane & 15;
    #pragma unroll
    for (int t = 0; t < 2; ++t) {
        const v8f& acc = t ? acc1 : acc0;
        const int col = n0 + wn * 32 + t * 16 + ln;
        #pragma unroll
        for (int v2 = 0; v2 < 8; ++v2) {
            const int row = m0 + wm * 16 + v2 + 8 * hh;
            float val = acc[v2];
            if (mode == 2) {
                size_t off = (size_t)row * DMODEL + col;
                out_f[off] = val + resid[off];
            } else {
                int bI = row / S, sI = row - bI * S;
                int h = col >> 6, dh = col & 63;
                size_t off = (mode == 0)
                    ? (((size_t)(bI * NHEAD + h) * S + sI) * 64 + dh)
                    : (((size_t)(bI * NHEAD + h) * 64 + dh) * S + sI);
                out_bf[off] = cvt1bf(val);
            }
        }
    }
}

// ---------------- Attention: 128 q-rows x one (b,h) per workgroup --------
// K/V tiles streamed by the Tensor Data Mover into double-buffered LDS.
#define TILE_E 4608          // 64 rows * 72 (padded) bf16 elements per tile
__global__ __launch_bounds__(256)
void attn_kernel(const unsigned short* __restrict__ Qh,
                 const unsigned short* __restrict__ Kh,
                 const unsigned short* __restrict__ VT,
                 float* __restrict__ ctx, float* __restrict__ attm, int S) {
    __shared__ unsigned short smem[TILE_E * 4 + 8 * 16 * 72];  // K0 K1 V0 V1 P
    unsigned short* Ps = smem + TILE_E * 4;
    const int tid = threadIdx.x, wave = tid >> 5, lane = tid & 31;
    const int hh = lane >> 4, ln = lane & 15;
    const int bh = blockIdx.y;
    const int q0 = blockIdx.x * 128;
    const unsigned short* Qg = Qh + (size_t)bh * S * 64;
    const unsigned short* Kg = Kh + (size_t)bh * S * 64;
    const unsigned short* Vg = VT + (size_t)bh * 64 * S;

    const unsigned int ldsK[2] = { (unsigned int)(size_t)(smem),
                                   (unsigned int)(size_t)(smem + TILE_E) };
    const unsigned int ldsV[2] = { (unsigned int)(size_t)(smem + 2 * TILE_E),
                                   (unsigned int)(size_t)(smem + 3 * TILE_E) };
    const u32x8 g1K = tdm_g1(64, (unsigned)S, 64, 64, 64ull);
    const u32x8 g1V = tdm_g1((unsigned)S, 64, 64, 64, (unsigned long long)S);

    // Q block 128x64 staged in smem[0 .. 2*TILE_E) (reused as K buffers after)
    for (int i = tid; i < 128 * 8; i += 256) {
        int r = i >> 3, c = (i & 7) * 8;
        *(uint4*)(smem + r * 72 + c) = *(const uint4*)(Qg + (size_t)(q0 + r) * 64 + c);
    }
    __syncthreads();
    Frag16 qa0, qa1;                           // loop-invariant A fragments
    frag_a(qa0, smem, 72, wave * 16, 0);
    frag_a(qa1, smem, 72, wave * 16, 32);
    __syncthreads();                           // Q region free for TDM reuse

    float mrow[8], lrow[8];
    #pragma unroll
    for (int v2 = 0; v2 < 8; ++v2) { mrow[v2] = -1e30f; lrow[v2] = 0.f; }

    // ---- pass 1: online row max / sum over all keys ----
    if (wave == 0) {
        tdm_load(tdm_g0(Kg, ldsK[0]), g1K);
        __builtin_amdgcn_s_wait_tensorcnt(0);
    }
    __syncthreads();
    int ib = 0;
    for (int kb = 0; kb < S; kb += 64) {
        if (wave == 0 && kb + 64 < S)          // prefetch next K tile
            tdm_load(tdm_g0(Kg + (size_t)(kb + 64) * 64, ldsK[ib ^ 1]), g1K);
        const unsigned short* Kc = smem + TILE_E * ib;
        v8f sc[4];
        #pragma unroll
        for (int c = 0; c < 4; ++c) {
            v8f a = {};
            Frag16 fb;
            frag_b(fb, Kc, 72, c * 16, 0);  a = wmma_bf16(qa0, fb, a);
            frag_b(fb, Kc, 72, c * 16, 32); a = wmma_bf16(qa1, fb, a);
            sc[c] = a;
        }
        #pragma unroll
        for (int v2 = 0; v2 < 8; ++v2) {
            float mx = -1e30f;
            #pragma unroll
            for (int c = 0; c < 4; ++c) mx = fmaxf(mx, sc[c][v2] * 0.125f);
            mx = fmaxf(mx, __shfl_xor(mx, 1)); mx = fmaxf(mx, __shfl_xor(mx, 2));
            mx = fmaxf(mx, __shfl_xor(mx, 4)); mx = fmaxf(mx, __shfl_xor(mx, 8));
            float mnew = fmaxf(mrow[v2], mx);
            float ssum = 0.f;
            #pragma unroll
            for (int c = 0; c < 4; ++c) ssum += __expf(sc[c][v2] * 0.125f - mnew);
            ssum += __shfl_xor(ssum, 1); ssum += __shfl_xor(ssum, 2);
            ssum += __shfl_xor(ssum, 4); ssum += __shfl_xor(ssum, 8);
            lrow[v2] = lrow[v2] * __expf(mrow[v2] - mnew) + ssum;
            mrow[v2] = mnew;
        }
        if (wave == 0) __builtin_amdgcn_s_wait_tensorcnt(0);
        __syncthreads();                       // publish prefetched tile
        ib ^= 1;
    }
    float inv_l[8];
    #pragma unroll
    for (int v2 = 0; v2 < 8; ++v2) inv_l[v2] = 1.0f / lrow[v2];

    // ---- pass 2: P = softmax, att_mean atomics, ctx = P x V ----
    v8f cx[4] = {};
    unsigned short* Pw = Ps + wave * 16 * 72;
    if (wave == 0) {
        tdm_load(tdm_g0(Kg, ldsK[0]), g1K);
        tdm_load(tdm_g0(Vg, ldsV[0]), g1V);
        __builtin_amdgcn_s_wait_tensorcnt(0);
    }
    __syncthreads();
    ib = 0;
    for (int kb = 0; kb < S; kb += 64) {
        if (wave == 0 && kb + 64 < S) {        // prefetch next K/V tiles
            tdm_load(tdm_g0(Kg + (size_t)(kb + 64) * 64, ldsK[ib ^ 1]), g1K);
            tdm_load(tdm_g0(Vg + (kb + 64),              ldsV[ib ^ 1]), g1V);
        }
        const unsigned short* Kc = smem + TILE_E * ib;
        const unsigned short* Vc = smem + TILE_E * (2 + ib);
        #pragma unroll
        for (int c = 0; c < 4; ++c) {
            v8f a = {};
            Frag16 fb;
            frag_b(fb, Kc, 72, c * 16, 0);  a = wmma_bf16(qa0, fb, a);
            frag_b(fb, Kc, 72, c * 16, 32); a = wmma_bf16(qa1, fb, a);
            #pragma unroll
            for (int v2 = 0; v2 < 8; ++v2) {
                float p = __expf(a[v2] * 0.125f - mrow[v2]) * inv_l[v2];
                int qrow = q0 + wave * 16 + v2 + 8 * hh;
                int kcol = kb + c * 16 + ln;
                atomicAdd(attm + (size_t)qrow * S + kcol, p * (1.0f / 32.0f));
                Pw[(v2 + 8 * hh) * 72 + c * 16 + ln] = cvt1bf(p);
            }
        }
        asm volatile("s_wait_dscnt 0" ::: "memory");  // P LDS write -> A-frag read
        #pragma unroll
        for (int c2 = 0; c2 < 4; ++c2) {
            Frag16 fa, fb;
            frag_a(fa, Pw, 72, 0, 0);  frag_b(fb, Vc, 72, c2 * 16, 0);  cx[c2] = wmma_bf16(fa, fb, cx[c2]);
            frag_a(fa, Pw, 72, 0, 32); frag_b(fb, Vc, 72, c2 * 16, 32); cx[c2] = wmma_bf16(fa, fb, cx[c2]);
        }
        if (wave == 0) __builtin_amdgcn_s_wait_tensorcnt(0);
        __syncthreads();
        ib ^= 1;
    }

    const int b = bh >> 4, h = bh & 15;
    #pragma unroll
    for (int c2 = 0; c2 < 4; ++c2)
        #pragma unroll
        for (int v2 = 0; v2 < 8; ++v2) {
            int srow = q0 + wave * 16 + v2 + 8 * hh;
            int col = h * 64 + c2 * 16 + ln;
            ctx[((size_t)b * S + srow) * DMODEL + col] = cx[c2][v2];
        }
}

// ---------------- LayerNorm (in place, one row per workgroup) ------------
__global__ __launch_bounds__(256)
void ln_kernel(float* __restrict__ y, const float* __restrict__ g,
               const float* __restrict__ beta) {
    float* x = y + (size_t)blockIdx.x * DMODEL;
    float vals[4], s = 0.f, s2 = 0.f;
    #pragma unroll
    for (int i = 0; i < 4; ++i) {
        vals[i] = x[threadIdx.x + 256 * i];
        s += vals[i]; s2 += vals[i] * vals[i];
    }
    for (int m = 1; m < 32; m <<= 1) { s += __shfl_xor(s, m); s2 += __shfl_xor(s2, m); }
    __shared__ float rs[8], rs2[8];
    const int wave = threadIdx.x >> 5, lane = threadIdx.x & 31;
    if (lane == 0) { rs[wave] = s; rs2[wave] = s2; }
    __syncthreads();
    s = rs[lane & 7]; s2 = rs2[lane & 7];
    for (int m = 1; m < 8; m <<= 1) { s += __shfl_xor(s, m); s2 += __shfl_xor(s2, m); }
    float mu  = s * (1.0f / DMODEL);
    float var = s2 * (1.0f / DMODEL) - mu * mu;
    float r = rsqrtf(var + 1e-5f);
    #pragma unroll
    for (int i = 0; i < 4; ++i) {
        int c = threadIdx.x + 256 * i;
        x[c] = (vals[i] - mu) * r * g[c] + beta[c];
    }
}

__global__ void zero_kernel(float4* p, int n4) {
    int i = blockIdx.x * 256 + threadIdx.x;
    if (i < n4) p[i] = make_float4(0.f, 0.f, 0.f, 0.f);
}

// ---------------- launcher ----------------
extern "C" void kernel_launch(void* const* d_in, const int* in_sizes, int n_in,
                              void* d_out, int out_size, void* d_ws, size_t ws_size,
                              hipStream_t stream) {
    (void)n_in; (void)out_size; (void)ws_size;
    const float* q    = (const float*)d_in[0];
    const float* k    = (const float*)d_in[1];
    const float* v    = (const float*)d_in[2];
    const float* Wq   = (const float*)d_in[3];
    const float* Wk   = (const float*)d_in[4];
    const float* Wv   = (const float*)d_in[5];
    const float* Wo   = (const float*)d_in[6];
    const float* ln_g = (const float*)d_in[7];
    const float* ln_b = (const float*)d_in[8];

    const int M = in_sizes[0] / DMODEL;   // B*S = 4096
    const int S = 2048;
    const int B = M / S;

    unsigned short* qh = (unsigned short*)d_ws;
    unsigned short* kh = qh + (size_t)M * DMODEL;
    unsigned short* vt = kh + (size_t)M * DMODEL;
    float* ctx = (float*)(vt + (size_t)M * DMODEL);

    float* y    = (float*)d_out;
    float* attm = y + (size_t)M * DMODEL;

    dim3 gg(DMODEL / 64, M / 64);
    gemm_xwT<<<gg, 256, 0, stream>>>(q, Wq, qh, nullptr, nullptr, S, 0);
    gemm_xwT<<<gg, 256, 0, stream>>>(k, Wk, kh, nullptr, nullptr, S, 0);
    gemm_xwT<<<gg, 256, 0, stream>>>(v, Wv, vt, nullptr, nullptr, S, 1);

    int n4 = (S * S) / 4;
    zero_kernel<<<(n4 + 255) / 256, 256, 0, stream>>>((float4*)attm, n4);

    attn_kernel<<<dim3(S / 128, B * NHEAD), 256, 0, stream>>>(qh, kh, vt, ctx, attm, S);

    gemm_xwT<<<gg, 256, 0, stream>>>(ctx, Wo, nullptr, y, q, S, 2);
    ln_kernel<<<M, 256, 0, stream>>>(y, ln_g, ln_b);
}
